// TransformerEncoderLayer_70832600645837
// MI455X (gfx1250) — compile-verified, matrix operand order checked by reading, not run on hardware
//
#include <hip/hip_runtime.h>

// ---------------------------------------------------------------------------
// Transformer encoder layer for MI455X (gfx1250), wave32.
// All GEMMs use v_wmma_f32_16x16x32_bf16 with bf16 operands kept k-contiguous
// (A:[M][K], B:[N][K]) so LDS staging is pure 16B copies done with the CDNA5
// async-to-LDS path (global_load_async_to_lds_b128 + ASYNCcnt), double-buffered.
// ---------------------------------------------------------------------------

#define F_BIAS   1
#define F_RELU   2
#define F_RES    4
#define F_ALPHA  8
#define F_OUTF32 16   // write fp32 C
#define F_OUTBF  32   // write bf16 copy Cb (same [M][N] layout)
#define F_VT     64   // write bf16 V-transpose copy Cb[b][h][n][k] (S=2048, dk=64)

typedef __attribute__((ext_vector_type(16))) __bf16 bf16x16;
typedef __attribute__((ext_vector_type(8)))  float  v8f;

union FragU {
    bf16x16 bf;
    uint4   u[2];
};

__device__ __forceinline__ unsigned short f2bf(float f) {
    union { __bf16 h; unsigned short s; } cv;
    cv.h = (__bf16)f;
    return cv.s;
}

// 16-byte async copy global -> LDS (ASYNCcnt, no VGPR round trip).
__device__ __forceinline__ void asyncCp16(const void* gp, void* lp) {
    unsigned           l = (unsigned)(unsigned long long)lp;   // LDS offset (low 32b)
    unsigned long long g = (unsigned long long)gp;
    asm volatile("global_load_async_to_lds_b128 %0, %1, off" :: "v"(l), "v"(g) : "memory");
}
__device__ __forceinline__ void waitAsync0() {
    asm volatile("s_wait_asynccnt 0x0" ::: "memory");
}

constexpr int BM = 128, BN = 128, BK = 32;
constexpr int SSTR = BK + 8;   // LDS row stride in halves (80B, 16B-aligned, skewed)

// ---------------------------------------------------------------------------
// Tiled WMMA GEMM.  A: [M][K] bf16 (or fp32 when !ABF16)   B: [N][K] bf16
// Batched via blockIdx.z -> (zb=z/nH, zh=z%nH) element offsets.
// 256 threads = 8 waves; wave (wm 0..3, wn 0..1) owns a 32x64 sub-tile.
// M % 128 == 0 and K % 32 == 0 for every launch; GUARD handles N < BN.
// ---------------------------------------------------------------------------
template<int FLAGS, bool ABF16, bool GUARD>
__global__ __launch_bounds__(256) void gemm_bf16(
    const void* __restrict__ Av, const unsigned short* __restrict__ Bt,
    const float* __restrict__ bias, const float* __restrict__ Res,
    float* __restrict__ C, unsigned short* __restrict__ Cb,
    int M, int N, int K, int lda, int ldb, int ldc,
    float alpha, int nH,
    long long sAb, long long sAh, long long sBb, long long sBh,
    long long sCb, long long sCh)
{
    __shared__ __align__(16) unsigned short sA[2][BM][SSTR];
    __shared__ __align__(16) unsigned short sB[2][BN][SSTR];

    const int tid  = threadIdx.x;
    const int wave = tid >> 5;
    const int lane = tid & 31;
    const int lo   = lane & 15;
    const int hi   = lane >> 4;
    const int wm   = wave & 3;
    const int wn   = wave >> 2;

    const int mBase = blockIdx.y * BM;
    const int nBase = blockIdx.x * BN;

    const unsigned short* Ab = (const unsigned short*)Av;
    const float*          Af = (const float*)Av;
    {
        const int z  = blockIdx.z;
        const int zb = z / nH, zh = z % nH;
        Ab += zb * sAb + zh * sAh;
        Af += zb * sAb + zh * sAh;
        Bt += zb * sBb + zh * sBh;
        C  += zb * sCb + zh * sCh;
        Cb += zb * sCb + zh * sCh;
    }

    // ---- staging helpers -------------------------------------------------
    auto stageA_async = [&](int kt, int buf) {
#pragma unroll
        for (int it = 0; it < 2; ++it) {            // 512 16B chunks: 128 rows x 4
            int idx = it * 256 + tid;
            int row = idx >> 2;
            int cv  = idx & 3;
            asyncCp16(Ab + (long long)(mBase + row) * lda + kt * BK + cv * 8,
                      &sA[buf][row][cv * 8]);
        }
    };
    auto stageB_async = [&](int kt, int buf) {
#pragma unroll
        for (int it = 0; it < 2; ++it) {
            int idx = it * 256 + tid;
            int row = idx >> 2;
            int cv  = idx & 3;
            int gn  = nBase + row;
            if (!GUARD || gn < N) {
                asyncCp16(Bt + (long long)gn * ldb + kt * BK + cv * 8,
                          &sB[buf][row][cv * 8]);
            } else {
                *(uint4*)(&sB[buf][row][cv * 8]) = make_uint4(0u, 0u, 0u, 0u);
            }
        }
    };
    auto loadA_f32 = [&](int kt, float4* ra) {
#pragma unroll
        for (int it = 0; it < 4; ++it) {            // 1024 float4s: 128 rows x 8
            int idx = it * 256 + tid;
            int row = idx >> 3;
            int kv  = idx & 7;
            ra[it] = *(const float4*)(Af + (long long)(mBase + row) * lda + kt * BK + kv * 4);
        }
    };
    auto storeA_f32 = [&](const float4* ra, int buf) {
#pragma unroll
        for (int it = 0; it < 4; ++it) {
            int idx = it * 256 + tid;
            int row = idx >> 3;
            int kv  = idx & 7;
            ushort4 h;
            h.x = f2bf(ra[it].x); h.y = f2bf(ra[it].y);
            h.z = f2bf(ra[it].z); h.w = f2bf(ra[it].w);
            *(ushort4*)(&sA[buf][row][kv * 4]) = h;
        }
    };

    const v8f vzero = {0.f, 0.f, 0.f, 0.f, 0.f, 0.f, 0.f, 0.f};
    v8f acc[2][4];
#pragma unroll
    for (int mi = 0; mi < 2; ++mi)
#pragma unroll
        for (int ni = 0; ni < 4; ++ni)
            acc[mi][ni] = vzero;

    const int kTiles = (K + BK - 1) / BK;

    // ---- prologue: stage tile 0 into buffer 0 ----
    if constexpr (ABF16) {
        stageA_async(0, 0);
    } else {
        float4 r0[4];
        loadA_f32(0, r0);
        storeA_f32(r0, 0);
    }
    stageB_async(0, 0);

    // ---- main pipeline: compute buf[t&1] while staging buf[(t+1)&1] ----
    for (int t = 0; t < kTiles; ++t) {
        const int cur = t & 1;
        const int nxt = cur ^ 1;
        waitAsync0();
        __syncthreads();

        float4 ra[4];
        const bool haveNext = (t + 1 < kTiles);
        if (haveNext) {
            if constexpr (ABF16) stageA_async(t + 1, nxt);
            else                 loadA_f32(t + 1, ra);
            stageB_async(t + 1, nxt);
        }

        // fragment loads (ISA 7.12.2 16-bit 16x32 layout): two b128 each
        FragU aF[2], bF[4];
#pragma unroll
        for (int mi = 0; mi < 2; ++mi) {
            int r = wm * 32 + mi * 16 + lo;
            aF[mi].u[0] = *(const uint4*)(&sA[cur][r][hi * 8]);
            aF[mi].u[1] = *(const uint4*)(&sA[cur][r][16 + hi * 8]);
        }
#pragma unroll
        for (int ni = 0; ni < 4; ++ni) {
            int c = wn * 64 + ni * 16 + lo;
            bF[ni].u[0] = *(const uint4*)(&sB[cur][c][hi * 8]);
            bF[ni].u[1] = *(const uint4*)(&sB[cur][c][16 + hi * 8]);
        }
#pragma unroll
        for (int mi = 0; mi < 2; ++mi)
#pragma unroll
            for (int ni = 0; ni < 4; ++ni)
                acc[mi][ni] = __builtin_amdgcn_wmma_f32_16x16x32_bf16(
                    false, aF[mi].bf, false, bF[ni].bf,
                    (short)0, acc[mi][ni], false, false);

        if (haveNext && !ABF16) storeA_f32(ra, nxt);
    }

    // ---- epilogue: C layout (VGPR e -> M = hi*8 + e, N = lo) ----
#pragma unroll
    for (int ni = 0; ni < 4; ++ni) {
        int gc = nBase + wn * 64 + ni * 16 + lo;
        if (GUARD && gc >= N) continue;
        float bval = (FLAGS & F_BIAS) ? bias[gc] : 0.f;
#pragma unroll
        for (int mi = 0; mi < 2; ++mi) {
#pragma unroll
            for (int e = 0; e < 8; ++e) {
                int gr = mBase + wm * 32 + mi * 16 + hi * 8 + e;
                float v = acc[mi][ni][e];
                if (FLAGS & F_ALPHA) v *= alpha;
                v += bval;
                if (FLAGS & F_RELU) v = fmaxf(v, 0.f);
                long long off = (long long)gr * ldc + gc;
                if (FLAGS & F_RES) v += Res[off];
                if (FLAGS & F_OUTF32) C[off] = v;
                if (FLAGS & F_OUTBF)  Cb[off] = f2bf(v);
                if (FLAGS & F_VT) {
                    // V-projection: gr = token (b = gr/2048, k = gr%2048),
                    // gc = feature (h = gc/64, n = gc%64) -> Vt[b][h][n][k]
                    int b  = gr >> 11, k = gr & 2047;
                    int h2 = gc >> 6,  n2 = gc & 63;
                    Cb[((((long long)b * 12 + h2) * 64 + n2) << 11) | k] = f2bf(v);
                }
            }
        }
    }
}

// ---------------------------------------------------------------------------
// Weight transpose + fp32->bf16:  W[K][N] -> Wt[N][K]  (K,N multiples of 32)
// ---------------------------------------------------------------------------
__global__ __launch_bounds__(256) void transpose_cvt(
    const float* __restrict__ W, unsigned short* __restrict__ Wt, int K, int N)
{
    __shared__ float tile[32][33];
    const int n0 = blockIdx.x * 32, k0 = blockIdx.y * 32;
    const int tx = threadIdx.x & 31, ty = threadIdx.x >> 5;  // ty: 0..7
#pragma unroll
    for (int i = ty; i < 32; i += 8)
        tile[i][tx] = W[(long long)(k0 + i) * N + n0 + tx];
    __syncthreads();
#pragma unroll
    for (int i = ty; i < 32; i += 8)
        Wt[(long long)(n0 + i) * K + k0 + tx] = f2bf(tile[tx][i]);
}

// ---------------------------------------------------------------------------
// fp32 -> bf16 elementwise (n4 = count/4)
// ---------------------------------------------------------------------------
__global__ __launch_bounds__(256) void cvt_bf16(
    const float* __restrict__ in, unsigned short* __restrict__ out)
{
    long long i = (long long)blockIdx.x * 256 + threadIdx.x;
    float4 v = ((const float4*)in)[i];
    ushort4 h;
    h.x = f2bf(v.x); h.y = f2bf(v.y); h.z = f2bf(v.z); h.w = f2bf(v.w);
    ((ushort4*)out)[i] = h;
}

// ---------------------------------------------------------------------------
// Row softmax over 2048-wide rows, register resident (1 read, 1 write).
// ---------------------------------------------------------------------------
__global__ __launch_bounds__(256) void softmax2048(float* __restrict__ attn)
{
    const long long row = blockIdx.x;
    float* p = attn + row * 2048;
    const int tid = threadIdx.x;

    float4 a = ((const float4*)p)[tid * 2];
    float4 b = ((const float4*)p)[tid * 2 + 1];

    __shared__ float red[256];
    float m = fmaxf(fmaxf(fmaxf(a.x, a.y), fmaxf(a.z, a.w)),
                    fmaxf(fmaxf(b.x, b.y), fmaxf(b.z, b.w)));
    red[tid] = m; __syncthreads();
#pragma unroll
    for (int s = 128; s > 0; s >>= 1) {
        if (tid < s) red[tid] = fmaxf(red[tid], red[tid + s]);
        __syncthreads();
    }
    m = red[0]; __syncthreads();

    a.x = __expf(a.x - m); a.y = __expf(a.y - m);
    a.z = __expf(a.z - m); a.w = __expf(a.w - m);
    b.x = __expf(b.x - m); b.y = __expf(b.y - m);
    b.z = __expf(b.z - m); b.w = __expf(b.w - m);

    red[tid] = (a.x + a.y + a.z + a.w) + (b.x + b.y + b.z + b.w);
    __syncthreads();
#pragma unroll
    for (int s = 128; s > 0; s >>= 1) {
        if (tid < s) red[tid] += red[tid + s];
        __syncthreads();
    }
    float inv = 1.f / red[0];

    a.x *= inv; a.y *= inv; a.z *= inv; a.w *= inv;
    b.x *= inv; b.y *= inv; b.z *= inv; b.w *= inv;
    ((float4*)p)[tid * 2]     = a;
    ((float4*)p)[tid * 2 + 1] = b;
}

// ---------------------------------------------------------------------------
// LayerNorm over 768-wide rows; optional bf16 side copy.
// ---------------------------------------------------------------------------
__global__ __launch_bounds__(256) void layernorm768(
    const float* __restrict__ in, const float* __restrict__ g,
    const float* __restrict__ b, float* __restrict__ out,
    unsigned short* __restrict__ outb)
{
    const long long row = blockIdx.x;
    const float* p = in + row * 768;
    const int tid = threadIdx.x;

    float v0 = p[tid], v1 = p[tid + 256], v2 = p[tid + 512];

    __shared__ float red[256];
    red[tid] = v0 + v1 + v2; __syncthreads();
#pragma unroll
    for (int s = 128; s > 0; s >>= 1) {
        if (tid < s) red[tid] += red[tid + s];
        __syncthreads();
    }
    float mu = red[0] * (1.f / 768.f); __syncthreads();

    float d0 = v0 - mu, d1 = v1 - mu, d2 = v2 - mu;
    red[tid] = d0 * d0 + d1 * d1 + d2 * d2; __syncthreads();
#pragma unroll
    for (int s = 128; s > 0; s >>= 1) {
        if (tid < s) red[tid] += red[tid + s];
        __syncthreads();
    }
    float rs = rsqrtf(red[0] * (1.f / 768.f) + 1e-5f);

    float o0 = d0 * rs * g[tid]       + b[tid];
    float o1 = d1 * rs * g[tid + 256] + b[tid + 256];
    float o2 = d2 * rs * g[tid + 512] + b[tid + 512];

    float* q = out + row * 768;
    q[tid] = o0; q[tid + 256] = o1; q[tid + 512] = o2;
    if (outb) {
        unsigned short* qb = outb + row * 768;
        qb[tid] = f2bf(o0); qb[tid + 256] = f2bf(o1); qb[tid + 512] = f2bf(o2);
    }
}

// ---------------------------------------------------------------------------
extern "C" void kernel_launch(void* const* d_in, const int* in_sizes, int n_in,
                              void* d_out, int out_size, void* d_ws, size_t ws_size,
                              hipStream_t stream)
{
    (void)in_sizes; (void)n_in; (void)out_size; (void)ws_size;

    const float* x   = (const float*)d_in[0];
    const float* Wq  = (const float*)d_in[1];
    const float* bq  = (const float*)d_in[2];
    const float* Wk  = (const float*)d_in[3];
    const float* bk  = (const float*)d_in[4];
    const float* Wv  = (const float*)d_in[5];
    const float* bv  = (const float*)d_in[6];
    const float* Wo  = (const float*)d_in[7];
    const float* bo  = (const float*)d_in[8];
    const float* W1  = (const float*)d_in[9];
    const float* b1  = (const float*)d_in[10];
    const float* W2  = (const float*)d_in[11];
    const float* b2  = (const float*)d_in[12];
    const float* g1  = (const float*)d_in[13];
    const float* be1 = (const float*)d_in[14];
    const float* g2  = (const float*)d_in[15];
    const float* be2 = (const float*)d_in[16];

    float* out = (float*)d_out;
    const long long NT = 4096, D = 768, FF = 3072, S = 2048, H = 12;

    // ---- workspace carve-up (bf16 buffers first, then fp32) ----
    char* w = (char*)d_ws;
    auto takeU16 = [&](long long elems) { unsigned short* p = (unsigned short*)w; w += elems * 2; return p; };
    auto takeF32 = [&](long long elems) { float* p = (float*)w; w += elems * 4; return p; };

    unsigned short* xb   = takeU16(NT * D);
    unsigned short* Qb   = takeU16(NT * D);
    unsigned short* Kb   = takeU16(NT * D);
    unsigned short* Vt   = takeU16(2 * H * 64 * S);   // [b][h][n][k]
    unsigned short* ctxb = takeU16(NT * D);
    unsigned short* x1b  = takeU16(NT * D);
    unsigned short* hb   = takeU16(NT * FF);
    unsigned short* Wqt  = takeU16(D * D);
    unsigned short* Wkt  = takeU16(D * D);
    unsigned short* Wvt  = takeU16(D * D);
    unsigned short* Wot  = takeU16(D * D);
    unsigned short* W1t  = takeU16(D * FF);           // [3072][768]
    unsigned short* W2t  = takeU16(FF * D);           // [768][3072]
    float*          T1   = takeF32(NT * D);
    float*          X1   = takeF32(NT * D);

    float* x2out = out;          // [4096 x 768]
    float* attn  = out + NT * D; // [2,12,2048,2048]

    dim3 blk(256);

    // ---- one-time conversions (cheap) ----
    transpose_cvt<<<dim3(24, 24), blk, 0, stream>>>(Wq, Wqt, 768, 768);
    transpose_cvt<<<dim3(24, 24), blk, 0, stream>>>(Wk, Wkt, 768, 768);
    transpose_cvt<<<dim3(24, 24), blk, 0, stream>>>(Wv, Wvt, 768, 768);
    transpose_cvt<<<dim3(24, 24), blk, 0, stream>>>(Wo, Wot, 768, 768);
    transpose_cvt<<<dim3(96, 24), blk, 0, stream>>>(W1, W1t, 768, 3072);
    transpose_cvt<<<dim3(24, 96), blk, 0, stream>>>(W2, W2t, 3072, 768);
    cvt_bf16<<<dim3(3072), blk, 0, stream>>>(x, xb);

    // ---- QKV projections (bf16 outputs only; V written head-transposed) ----
    gemm_bf16<F_BIAS | F_OUTBF, true, false><<<dim3(6, 32, 1), blk, 0, stream>>>(
        xb, Wqt, bq, nullptr, nullptr, Qb, 4096, 768, 768, 768, 768, 768, 1.f, 1, 0,0,0,0,0,0);
    gemm_bf16<F_BIAS | F_OUTBF, true, false><<<dim3(6, 32, 1), blk, 0, stream>>>(
        xb, Wkt, bk, nullptr, nullptr, Kb, 4096, 768, 768, 768, 768, 768, 1.f, 1, 0,0,0,0,0,0);
    gemm_bf16<F_BIAS | F_VT, true, false><<<dim3(6, 32, 1), blk, 0, stream>>>(
        xb, Wvt, bv, nullptr, nullptr, Vt, 4096, 768, 768, 768, 768, 768, 1.f, 1, 0,0,0,0,0,0);

    // ---- scores = Q @ K^T * 0.125 per (b,h); both operands bf16 [rows][64] ----
    gemm_bf16<F_ALPHA | F_OUTF32, true, false><<<dim3(16, 16, 24), blk, 0, stream>>>(
        Qb, Kb, nullptr, nullptr, attn, nullptr,
        2048, 2048, 64, 768, 768, 2048, 0.125f, 12,
        S * D, 64, S * D, 64, H * S * S, S * S);

    // ---- softmax in place in d_out ----
    softmax2048<<<dim3(24 * 2048), blk, 0, stream>>>(attn);

    // ---- ctx = attn(fp32) @ Vt(bf16 [n][k]) per (b,h); bf16 output only ----
    gemm_bf16<F_OUTBF, false, true><<<dim3(1, 16, 24), blk, 0, stream>>>(
        attn, Vt, nullptr, nullptr, nullptr, ctxb,
        2048, 64, 2048, 2048, 2048, 768, 1.f, 12,
        H * S * S, S * S, (long long)H * 64 * S, 64 * S, S * D, 64);

    // ---- attn_out = ctx @ Wo + bo + x (residual fused) -> T1 fp32 ----
    gemm_bf16<F_BIAS | F_RES | F_OUTF32, true, false><<<dim3(6, 32, 1), blk, 0, stream>>>(
        ctxb, Wot, bo, x, T1, nullptr, 4096, 768, 768, 768, 768, 768, 1.f, 1, 0,0,0,0,0,0);

    // ---- x1 = LN(T1) (fp32 + bf16 copy) ----
    layernorm768<<<dim3(4096), blk, 0, stream>>>(T1, g1, be1, X1, x1b);

    // ---- h = relu(x1 @ W1 + b1) -> bf16 only ----
    gemm_bf16<F_BIAS | F_RELU | F_OUTBF, true, false><<<dim3(24, 32, 1), blk, 0, stream>>>(
        x1b, W1t, b1, nullptr, nullptr, hb, 4096, 3072, 768, 768, 768, 3072, 1.f, 1, 0,0,0,0,0,0);

    // ---- y2 = h @ W2 + b2 + x1 -> T1 fp32 ----
    gemm_bf16<F_BIAS | F_RES | F_OUTF32, true, false><<<dim3(6, 32, 1), blk, 0, stream>>>(
        hb, W2t, b2, X1, T1, nullptr, 4096, 768, 3072, 3072, 3072, 768, 1.f, 1, 0,0,0,0,0,0);

    // ---- x2 = LN(T1) -> d_out ----
    layernorm768<<<dim3(4096), blk, 0, stream>>>(T1, g2, be2, x2out, nullptr);
}
